// eRPE_86870008529292
// MI455X (gfx1250) — matrix operand order repaired and checked by reading, not compiled, not run on hardware
//
#include <hip/hip_runtime.h>
#include <hip/hip_bf16.h>
#include <math.h>

typedef __bf16 bf16_t;
typedef __attribute__((ext_vector_type(16))) __bf16 v16bf;
typedef __attribute__((ext_vector_type(8)))  __bf16 v8bf;
typedef __attribute__((ext_vector_type(8)))  float  v8f;

#define WMMA_BF16(a, b, c) \
  __builtin_amdgcn_wmma_f32_16x16x32_bf16(false, (a), false, (b), (short)0, (c), false, false)

#define CAT16(lo, hi) \
  __builtin_shufflevector((lo), (hi), 0, 1, 2, 3, 4, 5, 6, 7, 8, 9, 10, 11, 12, 13, 14, 15)

constexpr int B_ = 32, L_ = 512, C_ = 512, H_ = 8, DH_ = 64;

// ---------------------------------------------------------------------------
// Kernel 0: fp32 -> bf16 conversion (x and weights)
// ---------------------------------------------------------------------------
__global__ __launch_bounds__(256) void f32_to_bf16_kernel(const float* __restrict__ src,
                                                          bf16_t* __restrict__ dst, int n) {
  int i = blockIdx.x * blockDim.x + threadIdx.x;
  if (i < n) dst[i] = (bf16_t)src[i];
}

// ---------------------------------------------------------------------------
// Kernel 1: QKV projection GEMM.  One wave -> one 32x64 output tile
// (two 16-row D halves sharing every B fragment; 8 accumulator chains).
// which 0: Q (B,H,L,DH) ; which 1: K (B,H,L,DH) ; which 2: V^T (B,H,DH,L)
// ---------------------------------------------------------------------------
__global__ __launch_bounds__(256) void qkv_gemm_kernel(
    const bf16_t* __restrict__ xbf, const bf16_t* __restrict__ wqb,
    const bf16_t* __restrict__ wkb, const bf16_t* __restrict__ wvb,
    bf16_t* __restrict__ Qo, bf16_t* __restrict__ Ko, bf16_t* __restrict__ Vto) {
  const int lane = threadIdx.x & 31;
  const int wave = blockIdx.x * (blockDim.x >> 5) + (threadIdx.x >> 5);
  const int which   = wave >> 12;   // 0..2   (4096 waves each)
  const int rem     = wave & 4095;
  const int rowTile = rem >> 3;     // 0..511 (32-row tiles of B*L)
  const int cg      = rem & 7;      // 64-col group == head index

  const bf16_t* W = (which == 0) ? wqb : (which == 1) ? wkb : wvb;

  const int hs = lane >> 4;   // half-select
  const int ln = lane & 15;
  const bf16_t* xr0 = xbf + (size_t)(rowTile * 32 + ln) * C_;       // rows 0-15
  const bf16_t* xr1 = xr0 + (size_t)16 * C_;                        // rows 16-31

  v8f accA[4] = {};   // rows 0-15  x cols t*16..t*16+15
  v8f accB[4] = {};   // rows 16-31 x cols t*16..t*16+15
  for (int c0 = 0; c0 < C_; c0 += 32) {
    // A fragments: two 16x32 bf16 tiles of x (contiguous k = 16*hs + s order)
    v16bf a0 = CAT16(*(const v8bf*)(xr0 + c0 + hs * 16),
                     *(const v8bf*)(xr0 + c0 + hs * 16 + 8));
    v16bf a1 = CAT16(*(const v8bf*)(xr1 + c0 + hs * 16),
                     *(const v8bf*)(xr1 + c0 + hs * 16 + 8));
    const bf16_t* wb = W + c0 + hs * 16;
#pragma unroll
    for (int t = 0; t < 4; ++t) {
      // B fragment: column j of W^T == row j of W (contiguous 32B load)
      v16bf b = *(const v16bf*)(wb + (size_t)(cg * 64 + t * 16 + ln) * C_);
      accA[t] = WMMA_BF16(a0, b, accA[t]);
      accB[t] = WMMA_BF16(a1, b, accB[t]);
    }
  }

  const int b  = rowTile >> 4;           // batch (16 x 32-row tiles per batch)
  const int lb = (rowTile & 15) * 32;    // sequence base within batch
  const size_t bhc = (size_t)b * H_ + cg;
#pragma unroll
  for (int g = 0; g < 2; ++g) {
#pragma unroll
    for (int t = 0; t < 4; ++t) {
      const v8f acc = g ? accB[t] : accA[t];
      const int dh = t * 16 + ln;
      const int l0 = lb + g * 16 + hs * 8;   // D layout: rows hs*8..hs*8+7
      if (which == 2) {
        // V^T: lane's 8 rows are contiguous along L -> one 16B store
        v8bf pk;
#pragma unroll
        for (int r = 0; r < 8; ++r) pk[r] = (bf16_t)acc[r];
        *(v8bf*)(Vto + (bhc * DH_ + dh) * L_ + l0) = pk;
      } else {
        bf16_t* dst = (which == 0) ? Qo : Ko;
#pragma unroll
        for (int r = 0; r < 8; ++r)
          dst[(bhc * L_ + l0 + r) * DH_ + dh] = (bf16_t)acc[r];
      }
    }
  }
}

// ---------------------------------------------------------------------------
// Kernel 2: attention. One wave per (b, h, 16-row q block).
// S = (Q K^T) * scale -> LDS fp32 16x512 -> softmax -> +bias -> bf16 P -> P V
// ---------------------------------------------------------------------------
__global__ __launch_bounds__(32) void attn_kernel(
    const bf16_t* __restrict__ Qo, const bf16_t* __restrict__ Ko,
    const bf16_t* __restrict__ Vt, const float* __restrict__ bias_table,
    float* __restrict__ out_pre) {
  __shared__ float  S[16][512];     // 32 KB fp32 scores
  __shared__ bf16_t P[16][512];     // 16 KB bf16 probabilities (+bias)
  const int lane = threadIdx.x;
  const int ln = lane & 15, hs = lane >> 4;
  const int qt = blockIdx.x, h = blockIdx.y, b = blockIdx.z;
  const int q0 = qt * 16;
  const size_t bh = (size_t)b * H_ + h;
  const float scale = 0.044194173824159216f;      // C^-0.5 = 1/sqrt(512)

  // Q fragments (16 rows x 64 dh -> two 16x32 A tiles)
  const bf16_t* qr = Qo + (bh * L_ + q0 + ln) * DH_;
  v16bf a0 = CAT16(*(const v8bf*)(qr + hs * 16),
                   *(const v8bf*)(qr + hs * 16 + 8));
  v16bf a1 = CAT16(*(const v8bf*)(qr + 32 + hs * 16),
                   *(const v8bf*)(qr + 40 + hs * 16));

  // Pull the V^T tile (64 KB) toward cache while we do scores + softmax.
  const bf16_t* vbase = Vt + bh * DH_ * L_;
#pragma unroll
  for (int i = 0; i < 16; ++i)
    __builtin_prefetch((const char*)vbase + (size_t)(i * 32 + lane) * 128, 0, 1);

  // scores: 32 column tiles of 16 k-indices, K-dim = 64 (2 WMMAs each)
  for (int kt = 0; kt < 32; ++kt) {
    const bf16_t* kr = Ko + (bh * L_ + kt * 16 + ln) * DH_;  // B col = k row of K
    v16bf b0 = *(const v16bf*)(kr + hs * 16);
    v16bf b1 = *(const v16bf*)(kr + 32 + hs * 16);
    v8f c = {};
    c = WMMA_BF16(a0, b0, c);
    c = WMMA_BF16(a1, b1, c);
    const int col = kt * 16 + ln;
#pragma unroll
    for (int r = 0; r < 8; ++r) S[hs * 8 + r][col] = c[r] * scale;
  }
  __syncthreads();

  // per-row softmax, then post-softmax relative bias (rel_index = k - q + 511),
  // written out as bf16 P ready for WMMA.
  for (int m = 0; m < 16; ++m) {
    float mx = -1e30f;
    for (int c = lane; c < 512; c += 32) mx = fmaxf(mx, S[m][c]);
    for (int off = 16; off > 0; off >>= 1) mx = fmaxf(mx, __shfl_xor(mx, off, 32));
    float sum = 0.f;
    for (int c = lane; c < 512; c += 32) {
      float e = __expf(S[m][c] - mx);
      S[m][c] = e;
      sum += e;
    }
    for (int off = 16; off > 0; off >>= 1) sum += __shfl_xor(sum, off, 32);
    const float inv = 1.0f / sum;
    const int q = q0 + m;
    for (int c = lane; c < 512; c += 32)
      P[m][c] = (bf16_t)(S[m][c] * inv + bias_table[(c - q + (L_ - 1)) * H_ + h]);
  }
  __syncthreads();

  // out = P(16x512) @ V(512x64); A fragments = two 16B LDS vector loads,
  // B fragments = contiguous 32B loads from V^T.
  v8f o0 = {}, o1 = {}, o2 = {}, o3 = {};
  for (int kc = 0; kc < 16; ++kc) {
    const int k0 = kc * 32;
    v16bf aP = CAT16(*(const v8bf*)&P[ln][k0 + hs * 16],
                     *(const v8bf*)&P[ln][k0 + hs * 16 + 8]);
    const bf16_t* vb = vbase + k0 + hs * 16;
    o0 = WMMA_BF16(aP, *(const v16bf*)(vb + (size_t)(ln)      * L_), o0);
    o1 = WMMA_BF16(aP, *(const v16bf*)(vb + (size_t)(16 + ln) * L_), o1);
    o2 = WMMA_BF16(aP, *(const v16bf*)(vb + (size_t)(32 + ln) * L_), o2);
    o3 = WMMA_BF16(aP, *(const v16bf*)(vb + (size_t)(48 + ln) * L_), o3);
  }
  v8f oc[4] = {o0, o1, o2, o3};
  for (int t = 0; t < 4; ++t) {
#pragma unroll
    for (int r = 0; r < 8; ++r) {
      const int m = hs * 8 + r;
      out_pre[((size_t)b * L_ + q0 + m) * C_ + h * DH_ + t * 16 + ln] = oc[t][r];
    }
  }
}

// ---------------------------------------------------------------------------
// Kernel 3: LayerNorm over C=512 per (b,l) row.
// ---------------------------------------------------------------------------
__global__ __launch_bounds__(256) void ln_kernel(const float* __restrict__ xin,
                                                 const float* __restrict__ gamma,
                                                 const float* __restrict__ beta,
                                                 float* __restrict__ out) {
  __shared__ float red[256];
  const int row = blockIdx.x;
  const int t = threadIdx.x;
  const float* xr = xin + (size_t)row * C_;
  const float a = xr[t], bv = xr[t + 256];
  red[t] = a + bv;
  __syncthreads();
  for (int off = 128; off > 0; off >>= 1) {
    if (t < off) red[t] += red[t + off];
    __syncthreads();
  }
  const float mu = red[0] * (1.0f / 512.0f);
  __syncthreads();
  const float da = a - mu, db = bv - mu;
  red[t] = da * da + db * db;
  __syncthreads();
  for (int off = 128; off > 0; off >>= 1) {
    if (t < off) red[t] += red[t + off];
    __syncthreads();
  }
  const float inv = rsqrtf(red[0] * (1.0f / 512.0f) + 1e-5f);
  out[(size_t)row * C_ + t]       = gamma[t]       * da * inv + beta[t];
  out[(size_t)row * C_ + t + 256] = gamma[t + 256] * db * inv + beta[t + 256];
}

// ---------------------------------------------------------------------------
extern "C" void kernel_launch(void* const* d_in, const int* in_sizes, int n_in,
                              void* d_out, int out_size, void* d_ws, size_t ws_size,
                              hipStream_t stream) {
  (void)in_sizes; (void)n_in; (void)out_size; (void)ws_size;
  const float* x     = (const float*)d_in[0];
  const float* wq    = (const float*)d_in[1];
  const float* wk    = (const float*)d_in[2];
  const float* wv    = (const float*)d_in[3];
  const float* bias  = (const float*)d_in[4];
  const float* gamma = (const float*)d_in[5];
  const float* beta  = (const float*)d_in[6];
  // d_in[7] (rel_index) is recomputed analytically: k - q + (L-1)

  char* ws = (char*)d_ws;
  size_t off = 0;
  bf16_t* xbf = (bf16_t*)(ws + off); off += (size_t)B_ * L_ * C_ * 2;  // 16 MB
  bf16_t* wqb = (bf16_t*)(ws + off); off += (size_t)C_ * C_ * 2;
  bf16_t* wkb = (bf16_t*)(ws + off); off += (size_t)C_ * C_ * 2;
  bf16_t* wvb = (bf16_t*)(ws + off); off += (size_t)C_ * C_ * 2;
  bf16_t* Qb  = (bf16_t*)(ws + off); off += (size_t)B_ * L_ * C_ * 2;  // 16 MB
  bf16_t* Kb  = (bf16_t*)(ws + off); off += (size_t)B_ * L_ * C_ * 2;  // 16 MB
  bf16_t* Vtb = (bf16_t*)(ws + off); off += (size_t)B_ * L_ * C_ * 2;  // 16 MB
  float* out_pre = (float*)(ws + off); off += (size_t)B_ * L_ * C_ * 4; // 32 MB

  const int nx = B_ * L_ * C_;
  const int nw = C_ * C_;
  f32_to_bf16_kernel<<<(nx + 255) / 256, 256, 0, stream>>>(x, xbf, nx);
  f32_to_bf16_kernel<<<(nw + 255) / 256, 256, 0, stream>>>(wq, wqb, nw);
  f32_to_bf16_kernel<<<(nw + 255) / 256, 256, 0, stream>>>(wk, wkb, nw);
  f32_to_bf16_kernel<<<(nw + 255) / 256, 256, 0, stream>>>(wv, wvb, nw);

  // 3 outputs * 512 row tiles (32 rows) * 8 col groups = 12288 waves / 8 per block
  qkv_gemm_kernel<<<1536, 256, 0, stream>>>(xbf, wqb, wkb, wvb, Qb, Kb, Vtb);

  attn_kernel<<<dim3(L_ / 16, H_, B_), 32, 0, stream>>>(Qb, Kb, Vtb, bias, out_pre);

  ln_kernel<<<B_ * L_, 256, 0, stream>>>(out_pre, gamma, beta, (float*)d_out);
}
